// LocalGlobalMultiheadAttention_87814901334748
// MI455X (gfx1250) — matrix operand order repaired and checked by reading, not compiled, not run on hardware
//
#include <hip/hip_runtime.h>

typedef __attribute__((ext_vector_type(16))) __bf16          v16bf;
typedef __attribute__((ext_vector_type(8)))  float           v8f;
typedef __attribute__((ext_vector_type(4)))  float           v4f;
typedef __attribute__((ext_vector_type(16))) unsigned short  v16u;
typedef __attribute__((ext_vector_type(8)))  unsigned short  v8u;
typedef __attribute__((ext_vector_type(4)))  unsigned short  v4u;

#define BATCH   4
#define SEQ     2048
#define EMB     1024
#define HEADS   16
#define HD      64
#define LOCALK  128
#define SCALE   0.125f
#define MASKVAL (-3.0e38f)
#define MINIT   (-1.0e30f)

__device__ __forceinline__ unsigned short f2bf(float f) {
  unsigned u = __builtin_bit_cast(unsigned, f);
  u += 0x7fffu + ((u >> 16) & 1u);            // round-to-nearest-even
  return (unsigned short)(u >> 16);
}

__device__ __forceinline__ v16u pack16(v8u lo, v8u hi) {
  v16u r;
#pragma unroll
  for (int i = 0; i < 8; i++) { r[i] = lo[i]; r[i + 8] = hi[i]; }
  return r;
}

__device__ __forceinline__ v8f zero8() {
  v8f r;
#pragma unroll
  for (int i = 0; i < 8; i++) r[i] = 0.0f;
  return r;
}

__device__ __forceinline__ v8f wmma_bf16(v16u a, v16u b, v8f c) {
  return __builtin_amdgcn_wmma_f32_16x16x32_bf16(
      false, __builtin_bit_cast(v16bf, a),
      false, __builtin_bit_cast(v16bf, b),
      (short)0, c, false, false);
}

// CDNA5 async global->LDS copy (ASYNCcnt-tracked), 16 bytes per lane.
__device__ __forceinline__ void async_copy_b128(unsigned short* lds_dst,
                                                const unsigned short* gsrc) {
  __attribute__((address_space(3))) unsigned short* l =
      (__attribute__((address_space(3))) unsigned short*)lds_dst;
  const __attribute__((address_space(1))) unsigned short* g =
      (const __attribute__((address_space(1))) unsigned short*)gsrc;
  asm volatile("global_load_async_to_lds_b128 %0, %1, off"
               :: "v"(l), "v"(g) : "memory");
}
__device__ __forceinline__ void wait_asynccnt0() {
  asm volatile("s_wait_asynccnt 0" ::: "memory");
}

__device__ __forceinline__ float redmax16(float v) {
#pragma unroll
  for (int m = 1; m < 16; m <<= 1) v = fmaxf(v, __shfl_xor(v, m, 32));
  return v;
}
__device__ __forceinline__ float redsum16(float v) {
#pragma unroll
  for (int m = 1; m < 16; m <<= 1) v += __shfl_xor(v, m, 32);
  return v;
}

// ---------------------------------------------------------------------------
// Kernel 1: Q/K/V projection.  y = x @ W^T + b, bf16 output in [B,H,n,D].
// grid (128, 8, 3)   block 256 (8 waves): 64(M) x 128(N) tile per WG.
// ---------------------------------------------------------------------------
__global__ __launch_bounds__(256)
void lga_qkv_proj(const float* __restrict__ x,
                  const float* __restrict__ wq, const float* __restrict__ bq,
                  const float* __restrict__ wk, const float* __restrict__ bk,
                  const float* __restrict__ wv, const float* __restrict__ bv,
                  unsigned short* __restrict__ qkv) {
  __shared__ unsigned short Xs[64][40];   // 64 x 32 bf16 (+pad), rows 80B
  __shared__ unsigned short Ws[128][40];

  const int tid  = threadIdx.x;
  const int lane = tid & 31, wid = tid >> 5;
  const int col  = lane & 15, hig = lane >> 4;
  const int mSub = wid & 3,  nGrp = wid >> 2;
  const int row0 = blockIdx.x * 64;
  const int col0 = blockIdx.y * 128;
  const int sel  = blockIdx.z;

  const float* Wm   = (sel == 0) ? wq : (sel == 1) ? wk : wv;
  const float* bias = (sel == 0) ? bq : (sel == 1) ? bk : bv;
  unsigned short* dst = qkv + (size_t)sel * ((size_t)BATCH * HEADS * SEQ * HD);

  v8f acc[4];
#pragma unroll
  for (int c = 0; c < 4; c++) acc[c] = zero8();

  for (int kk = 0; kk < EMB; kk += 32) {
    __syncthreads();
    {   // stage X tile 64x32 f32 -> bf16 (8 floats / thread)
      int r = tid >> 2, c4 = (tid & 3) * 8;
      const float* src = &x[(size_t)(row0 + r) * EMB + kk + c4];
      v4f a = *(const v4f*)src;
      v4f b = *(const v4f*)(src + 4);
      v4u pa, pb;
#pragma unroll
      for (int i = 0; i < 4; i++) { pa[i] = f2bf(a[i]); pb[i] = f2bf(b[i]); }
      *(v4u*)&Xs[r][c4]     = pa;
      *(v4u*)&Xs[r][c4 + 4] = pb;
      if (kk + 32 < EMB) __builtin_prefetch(src + 32, 0, 1);
    }
    {   // stage W tile 128x32 f32 -> bf16 (16 floats / thread)
      int rr = tid >> 1, c16 = (tid & 1) * 16;
      const float* src = &Wm[(size_t)(col0 + rr) * EMB + kk + c16];
#pragma unroll
      for (int q = 0; q < 4; q++) {
        v4f a = *(const v4f*)(src + q * 4);
        v4u p;
#pragma unroll
        for (int i = 0; i < 4; i++) p[i] = f2bf(a[i]);
        *(v4u*)&Ws[rr][c16 + q * 4] = p;
      }
      if (kk + 32 < EMB) __builtin_prefetch(src + 32, 0, 1);
    }
    __syncthreads();

    const int off0 = hig ? 8 : 0;
    const int base = hig ? 16 : 0;
    v16u afrag = pack16(*(const v8u*)&Xs[mSub * 16 + col][base],
                        *(const v8u*)&Xs[mSub * 16 + col][base + 8]);
    v16u bfr[4];
#pragma unroll
    for (int c = 0; c < 4; c++) {
      int nr = nGrp * 64 + c * 16 + col;
      bfr[c] = pack16(*(const v8u*)&Ws[nr][base],
                      *(const v8u*)&Ws[nr][base + 8]);
    }
#pragma unroll
    for (int c = 0; c < 4; c++) acc[c] = wmma_bf16(afrag, bfr[c], acc[c]);
  }

#pragma unroll
  for (int c = 0; c < 4; c++) {
    int oc = col0 + nGrp * 64 + c * 16 + col;
    float bv_ = bias[oc];
    int h = oc >> 6, d = oc & 63;
#pragma unroll
    for (int v = 0; v < 8; v++) {
      int orow = row0 + mSub * 16 + v + hig * 8;
      int bb = orow >> 11, m = orow & 2047;
      dst[(((size_t)bb * HEADS + h) * SEQ + m) * HD + d] = f2bf(acc[c][v] + bv_);
    }
  }
}

// ---------------------------------------------------------------------------
// Kernel 2: flash attention per (query-block, head, batch).
// grid (16, 16, 4)  block 256 (8 waves x 16 query rows = 128 rows / WG).
// ---------------------------------------------------------------------------
__global__ __launch_bounds__(256)
void lga_attn(const unsigned short* __restrict__ qw,
              const unsigned short* __restrict__ kw,
              const unsigned short* __restrict__ vw,
              unsigned short* __restrict__ ow) {
  __shared__ unsigned short Qs[128][72];     // 128 x 64 bf16 (+pad), rows 144B
  __shared__ unsigned short Ks[32][72];      // 32 keys x 64 d
  __shared__ unsigned short Vt[64][40];      // transposed: 64 d x 32 keys
  __shared__ unsigned short Ps[8][16][40];   // per-wave P tile 16x32

  const int tid  = threadIdx.x;
  const int lane = tid & 31, wid = tid >> 5;
  const int col  = lane & 15, hig = lane >> 4;
  const int qb = blockIdx.x * 128;
  const int h  = blockIdx.y;
  const int bb = blockIdx.z;
  const size_t hb = ((size_t)bb * HEADS + h) * SEQ * HD;

  // stage Q block (128 x 64 bf16) via async global->LDS DMA
#pragma unroll
  for (int p = 0; p < 4; p++) {
    int idx = p * 2048 + tid * 8;
    int r = idx >> 6, c = idx & 63;
    async_copy_b128(&Qs[r][c], &qw[hb + (size_t)(qb + r) * HD + c]);
  }
  wait_asynccnt0();
  __syncthreads();

  const int iw   = qb + wid * 16;
  const int off0 = hig ? 8 : 0;
  const int kb   = hig ? 16 : 0;
  const int qr   = wid * 16 + col;
  v16u aq0 = pack16(*(const v8u*)&Qs[qr][kb],      *(const v8u*)&Qs[qr][kb + 8]);
  v16u aq1 = pack16(*(const v8u*)&Qs[qr][32 + kb], *(const v8u*)&Qs[qr][40 + kb]);

  float m[8], l[8];
  v8f O[4];
#pragma unroll
  for (int v = 0; v < 8; v++) { m[v] = MINIT; l[v] = 0.0f; }
#pragma unroll
  for (int c = 0; c < 4; c++) O[c] = zero8();

  const bool isGlobal = (h < 1);
  int jStart = isGlobal ? 0   : (qb - 128 < 0 ? 0 : qb - 128);
  int jEnd   = isGlobal ? SEQ : (qb + 256 > SEQ ? SEQ : qb + 256);

  for (int j0 = jStart; j0 < jEnd; j0 += 32) {
    __syncthreads();
    {   // stage K via async DMA; V via registers with transpose
      int key = tid >> 3, c8 = (tid & 7) * 8;
      async_copy_b128(&Ks[key][c8], &kw[hb + (size_t)(j0 + key) * HD + c8]);
      v8u vv = *(const v8u*)&vw[hb + (size_t)(j0 + key) * HD + c8];
#pragma unroll
      for (int j = 0; j < 8; j++) Vt[c8 + j][key] = vv[j];
    }
    wait_asynccnt0();
    __syncthreads();

    bool act = isGlobal || ((j0 + 31 >= iw - LOCALK) && (j0 <= iw + 15 + LOCALK));
    if (act) {
      v16u bk0 = pack16(*(const v8u*)&Ks[col][kb],           *(const v8u*)&Ks[col][kb + 8]);
      v16u bk1 = pack16(*(const v8u*)&Ks[col][32 + kb],      *(const v8u*)&Ks[col][40 + kb]);
      v16u bk2 = pack16(*(const v8u*)&Ks[16 + col][kb],      *(const v8u*)&Ks[16 + col][kb + 8]);
      v16u bk3 = pack16(*(const v8u*)&Ks[16 + col][32 + kb], *(const v8u*)&Ks[16 + col][40 + kb]);
      v8f s0 = zero8(), s1 = zero8();
      s0 = wmma_bf16(aq0, bk0, s0);
      s0 = wmma_bf16(aq1, bk1, s0);
      s1 = wmma_bf16(aq0, bk2, s1);
      s1 = wmma_bf16(aq1, bk3, s1);

#pragma unroll
      for (int v = 0; v < 8; v++) {
        int i = iw + v + hig * 8;
        float x0 = s0[v] * SCALE;
        float x1 = s1[v] * SCALE;
        if (!isGlobal) {
          int d0 = i - (j0 + col);      if (d0 < 0) d0 = -d0;
          int d1 = i - (j0 + 16 + col); if (d1 < 0) d1 = -d1;
          x0 = (d0 > LOCALK) ? MASKVAL : x0;
          x1 = (d1 > LOCALK) ? MASKVAL : x1;
        }
        float rm   = redmax16(fmaxf(x0, x1));
        float mnew = fmaxf(m[v], rm);
        float corr = __expf(m[v] - mnew);
        float p0 = __expf(x0 - mnew);
        float p1 = __expf(x1 - mnew);
        float rs = redsum16(p0 + p1);
        l[v] = l[v] * corr + rs;
        m[v] = mnew;
        Ps[wid][v + hig * 8][col]      = f2bf(p0);
        Ps[wid][v + hig * 8][16 + col] = f2bf(p1);
#pragma unroll
        for (int c = 0; c < 4; c++) O[c][v] *= corr;
      }

      __builtin_amdgcn_wave_barrier();
      asm volatile("s_wait_dscnt 0" ::: "memory");   // intra-wave LDS RAW (P tile)

      v16u pf = pack16(*(const v8u*)&Ps[wid][col][kb],
                       *(const v8u*)&Ps[wid][col][kb + 8]);
      v16u bv0 = pack16(*(const v8u*)&Vt[col][kb],      *(const v8u*)&Vt[col][kb + 8]);
      v16u bv1 = pack16(*(const v8u*)&Vt[16 + col][kb], *(const v8u*)&Vt[16 + col][kb + 8]);
      v16u bv2 = pack16(*(const v8u*)&Vt[32 + col][kb], *(const v8u*)&Vt[32 + col][kb + 8]);
      v16u bv3 = pack16(*(const v8u*)&Vt[48 + col][kb], *(const v8u*)&Vt[48 + col][kb + 8]);
      O[0] = wmma_bf16(pf, bv0, O[0]);
      O[1] = wmma_bf16(pf, bv1, O[1]);
      O[2] = wmma_bf16(pf, bv2, O[2]);
      O[3] = wmma_bf16(pf, bv3, O[3]);
    }
  }

  // normalize and store bf16 in [B, n, H, D]
#pragma unroll
  for (int c = 0; c < 4; c++) {
#pragma unroll
    for (int v = 0; v < 8; v++) {
      int i = iw + v + hig * 8;
      int d = c * 16 + col;
      float oval = O[c][v] / l[v];
      ow[((size_t)(bb * SEQ + i) * HEADS + h) * HD + d] = f2bf(oval);
    }
  }
}

// ---------------------------------------------------------------------------
// Kernel 3: output projection.  out = attn @ Wo^T + bo, fp32 output.
// grid (128, 8)  block 256.
// ---------------------------------------------------------------------------
__global__ __launch_bounds__(256)
void lga_oproj(const unsigned short* __restrict__ a,   // [B*n, E] bf16
               const float* __restrict__ wo, const float* __restrict__ bo,
               float* __restrict__ out) {
  __shared__ unsigned short Xs[64][40];
  __shared__ unsigned short Ws[128][40];

  const int tid  = threadIdx.x;
  const int lane = tid & 31, wid = tid >> 5;
  const int col  = lane & 15, hig = lane >> 4;
  const int mSub = wid & 3,  nGrp = wid >> 2;
  const int row0 = blockIdx.x * 64;
  const int col0 = blockIdx.y * 128;

  v8f acc[4];
#pragma unroll
  for (int c = 0; c < 4; c++) acc[c] = zero8();

  for (int kk = 0; kk < EMB; kk += 32) {
    __syncthreads();
    {   // stage A tile 64x32 bf16 via async DMA (8 elems / thread)
      int r = tid >> 2, c8 = (tid & 3) * 8;
      async_copy_b128(&Xs[r][c8], &a[(size_t)(row0 + r) * EMB + kk + c8]);
    }
    {   // stage Wo tile 128x32 f32 -> bf16
      int rr = tid >> 1, c16 = (tid & 1) * 16;
      const float* src = &wo[(size_t)(col0 + rr) * EMB + kk + c16];
#pragma unroll
      for (int q = 0; q < 4; q++) {
        v4f t = *(const v4f*)(src + q * 4);
        v4u p;
#pragma unroll
        for (int i = 0; i < 4; i++) p[i] = f2bf(t[i]);
        *(v4u*)&Ws[rr][c16 + q * 4] = p;
      }
      if (kk + 32 < EMB) __builtin_prefetch(src + 32, 0, 1);
    }
    wait_asynccnt0();
    __syncthreads();

    const int off0 = hig ? 8 : 0;
    const int base = hig ? 16 : 0;
    v16u afrag = pack16(*(const v8u*)&Xs[mSub * 16 + col][base],
                        *(const v8u*)&Xs[mSub * 16 + col][base + 8]);
    v16u bfr[4];
#pragma unroll
    for (int c = 0; c < 4; c++) {
      int nr = nGrp * 64 + c * 16 + col;
      bfr[c] = pack16(*(const v8u*)&Ws[nr][base],
                      *(const v8u*)&Ws[nr][base + 8]);
    }
#pragma unroll
    for (int c = 0; c < 4; c++) acc[c] = wmma_bf16(afrag, bfr[c], acc[c]);
  }

#pragma unroll
  for (int c = 0; c < 4; c++) {
    int oc = col0 + nGrp * 64 + c * 16 + col;
    float bv_ = bo[oc];
#pragma unroll
    for (int v = 0; v < 8; v++) {
      int orow = row0 + mSub * 16 + v + hig * 8;
      out[(size_t)orow * EMB + oc] = acc[c][v] + bv_;
    }
  }
}

// ---------------------------------------------------------------------------
extern "C" void kernel_launch(void* const* d_in, const int* in_sizes, int n_in,
                              void* d_out, int out_size, void* d_ws, size_t ws_size,
                              hipStream_t stream) {
  const float* x  = (const float*)d_in[0];
  const float* wq = (const float*)d_in[1];
  const float* bq = (const float*)d_in[2];
  const float* wk = (const float*)d_in[3];
  const float* bk = (const float*)d_in[4];
  const float* wv = (const float*)d_in[5];
  const float* bv = (const float*)d_in[6];
  const float* wo = (const float*)d_in[7];
  const float* bo = (const float*)d_in[8];

  unsigned short* ws = (unsigned short*)d_ws;
  const size_t QKV = (size_t)BATCH * HEADS * SEQ * HD;   // 8,388,608 elems

  lga_qkv_proj<<<dim3(128, 8, 3), 256, 0, stream>>>(x, wq, bq, wk, bk, wv, bv, ws);
  lga_attn<<<dim3(16, 16, 4), 256, 0, stream>>>(ws, ws + QKV, ws + 2 * QKV, ws + 3 * QKV);
  lga_oproj<<<dim3(128, 8), 256, 0, stream>>>(ws + 3 * QKV, wo, bo, (float*)d_out);
}
